// Head_75007308858015
// MI455X (gfx1250) — matrix operand order repaired and checked by reading, not compiled, not run on hardware
//
#include <hip/hip_runtime.h>
#include <hip/hip_bf16.h>

// ---------------------------------------------------------------------------
// Causal single-head attention for MI455X (gfx1250, wave32).
//   B=8, T=4096, C=64, H=128, fp32 in/out.
// v2: block = 128 query rows (8 waves). K/V chunks are staged into LDS with
// async copies (global_load_async_to_lds_b128, ASYNCcnt), double-buffered, and
// shared by all 8 waves -> 8x less L2 traffic and WMMA feeds from ds_load_b128
// instead of stalling on global latency. Matrix math: v_wmma_f32_16x16x32_f16.
// ---------------------------------------------------------------------------

#define T_SEQ  4096
#define B_BATCH 8
#define C_EMBD 64
#define HD     128

typedef _Float16 v16h __attribute__((ext_vector_type(16)));
typedef _Float16 v8h  __attribute__((ext_vector_type(8)));
typedef float    v8f  __attribute__((ext_vector_type(8)));

union Frag16 {
    v16h v;
    v8h  h[2];
    unsigned u[8];
};

__device__ inline v8f zero8() {
    v8f r;
    #pragma unroll
    for (int i = 0; i < 8; ++i) r[i] = 0.0f;
    return r;
}

__device__ inline unsigned lds_off(const void* p) {
    // For LDS-aperture flat pointers, addr[31:0] is the wave-relative LDS
    // byte offset (ISA 10.2 aperture rules) -> usable as async VDST operand.
    return (unsigned)(unsigned long long)p;
}

__device__ inline void async_copy16(unsigned lds_addr, const void* gptr) {
    unsigned long long g = (unsigned long long)gptr;
    asm volatile("global_load_async_to_lds_b128 %0, %1, off"
                 :: "v"(lds_addr), "v"(g)
                 : "memory");
}

__device__ inline void wait_async0() {
    asm volatile("s_wait_asynccnt 0x0" ::: "memory");
}

// ---------------------------------------------------------------------------
// Kernel 1: QKV projection. One block per token (128 threads, h = tid).
// Q is pre-scaled by H^-0.5. V is written transposed: VT[b][h][t].
// ---------------------------------------------------------------------------
__global__ __launch_bounds__(128) void qkv_proj_kernel(
    const float* __restrict__ x,
    const float* __restrict__ Wq,
    const float* __restrict__ Wk,
    const float* __restrict__ Wv,
    _Float16* __restrict__ Qh,
    _Float16* __restrict__ Kh,
    _Float16* __restrict__ VTh)
{
    __shared__ float xs[C_EMBD];
    const int bt = blockIdx.x;       // 0 .. B*T-1
    const int h  = threadIdx.x;      // 0 .. 127
    if (h < C_EMBD) xs[h] = x[(size_t)bt * C_EMBD + h];
    __syncthreads();

    float q = 0.f, k = 0.f, v = 0.f;
    #pragma unroll
    for (int c = 0; c < C_EMBD; ++c) {
        const float xv = xs[c];
        q = fmaf(xv, Wq[c * HD + h], q);
        k = fmaf(xv, Wk[c * HD + h], k);
        v = fmaf(xv, Wv[c * HD + h], v);
    }
    const float scale = 0.08838834764831845f;   // 128^-0.5, folded into Q
    Qh[(size_t)bt * HD + h] = (_Float16)(q * scale);
    Kh[(size_t)bt * HD + h] = (_Float16)k;
    const int b = bt / T_SEQ;
    const int t = bt - b * T_SEQ;
    VTh[((size_t)b * HD + h) * T_SEQ + t] = (_Float16)v;
}

// ---------------------------------------------------------------------------
// Kernel 2: flash attention. Block = 128 q rows of one batch (8 waves x 16).
// S^T = K * Q^T per wave (softmax = per-lane scalars + one xor-16 shuffle),
// O^T += V^T * P^T. K/V chunks double-buffered in LDS via async copies.
// ---------------------------------------------------------------------------
__global__ __launch_bounds__(256) void flash_attn_kernel(
    const _Float16* __restrict__ Qh,
    const _Float16* __restrict__ Kh,
    const _Float16* __restrict__ VTh,
    float* __restrict__ out)
{
    __shared__ _Float16 sK[2][32 * HD];    // 2 x 8 KB: [s=32][h=128]
    __shared__ _Float16 sV[2][HD * 32];    // 2 x 8 KB: [h=128][s=32]

    const int tid  = threadIdx.x;
    const int lane = tid & 31;
    const int wave = tid >> 5;
    const int lo   = lane & 15;
    const bool hi  = lane >= 16;

    const int blocksPerBatch = T_SEQ / 128;              // 32
    const int b       = blockIdx.x / blocksPerBatch;
    const int blockQ0 = (blockIdx.x % blocksPerBatch) * 128;
    const int q0      = blockQ0 + wave * 16;             // this wave's 16 q rows

    const _Float16* Qb = Qh  + (size_t)b * T_SEQ * HD;
    const _Float16* Kb = Kh  + (size_t)b * T_SEQ * HD;
    const _Float16* Vb = VTh + (size_t)b * HD * T_SEQ;

    // ---- cooperative async stage of one 32-row K/V chunk into LDS buffer ----
    auto stage = [&](int sB, int buf) {
        #pragma unroll
        for (int j = 0; j < 2; ++j) {                    // K: 32 rows x 256 B
            const int idx = tid + j * 256;
            const int row = idx >> 4, seg = idx & 15;    // 16 B segments
            async_copy16(lds_off(&sK[buf][row * HD + seg * 8]),
                         Kb + (size_t)(sB + row) * HD + seg * 8);
        }
        #pragma unroll
        for (int j = 0; j < 2; ++j) {                    // V^T: 128 rows x 64 B
            const int idx = tid + j * 256;
            const int h = idx >> 2, seg = idx & 3;
            async_copy16(lds_off(&sV[buf][h * 32 + seg * 8]),
                         Vb + (size_t)h * T_SEQ + sB + seg * 8);
        }
    };

    // Q B-fragments (16-bit B 32x16): lane<16 -> k 0..15, lane>=16 -> k 16..31.
    v16h qf[4];
    #pragma unroll
    for (int kk = 0; kk < 4; ++kk) {
        const _Float16* p = Qb + (size_t)(q0 + lo) * HD + kk * 32 + (hi ? 16 : 0);
        qf[kk] = *(const v16h*)p;
    }

    v8f acc[8];
    #pragma unroll
    for (int i = 0; i < 8; ++i) acc[i] = zero8();
    float m = -1e30f, l = 0.f;

    const int sLimit  = q0 + 16;                 // this wave needs s <= q0+15
    const int nChunks = (blockQ0 + 128) / 32;    // block-uniform trip count

    stage(0, 0);
    for (int i = 0; i < nChunks; ++i) {
        wait_async0();                           // my share of chunk i copies done
        __syncthreads();                         // everyone's done; prev buf free
        if (i + 1 < nChunks) stage(i + 1, (i + 1) & 1);

        const int sBase = i * 32;
        if (sBase < sLimit) {                    // wave-uniform causal gate
            const _Float16* kb = &sK[i & 1][0];
            const _Float16* vb = &sV[i & 1][0];

            // ---- scores^T: two 16x16 tiles (s-subtiles), K-dim 128 = 4x32 ----
            v8f c0 = zero8(), c1 = zero8();
            #pragma unroll
            for (int kk = 0; kk < 4; ++kk) {
                Frag16 a0, a1;
                const _Float16* k0p = kb + lo * HD + kk * 32 + (hi ? 16 : 0);
                a0.h[0] = *(const v8h*)k0p;
                a0.h[1] = *(const v8h*)(k0p + 8);
                const _Float16* k1p = k0p + 16 * HD;
                a1.h[0] = *(const v8h*)k1p;
                a1.h[1] = *(const v8h*)(k1p + 8);
                c0 = __builtin_amdgcn_wmma_f32_16x16x32_f16(false, a0.v, false, qf[kk],
                                                            (short)0, c0, false, false);
                c1 = __builtin_amdgcn_wmma_f32_16x16x32_f16(false, a1.v, false, qf[kk],
                                                            (short)0, c1, false, false);
            }

            // ---- causal mask: (s = sBase + r + 8*hi (+16), q = q0 + lo) ----
            const int qg = q0 + lo;
            if (sBase + 31 > q0) {
                #pragma unroll
                for (int r = 0; r < 8; ++r) {
                    const int sg = sBase + r + (hi ? 8 : 0);
                    if (sg > qg)      c0[r] = -1e30f;
                    if (sg + 16 > qg) c1[r] = -1e30f;
                }
            }

            // ---- online softmax (per-lane scalar + one xor-16 combine) ----
            float tmax = -1e30f;
            #pragma unroll
            for (int r = 0; r < 8; ++r) {
                tmax = fmaxf(tmax, c0[r]);
                tmax = fmaxf(tmax, c1[r]);
            }
            tmax = fmaxf(tmax, __shfl_xor(tmax, 16, 32));
            const float mnew  = fmaxf(m, tmax);
            const float alpha = __expf(m - mnew);
            float rsum = 0.f;
            #pragma unroll
            for (int r = 0; r < 8; ++r) {
                c0[r] = __expf(c0[r] - mnew);
                c1[r] = __expf(c1[r] - mnew);
                rsum += c0[r] + c1[r];
            }
            rsum += __shfl_xor(rsum, 16, 32);
            l = l * alpha + rsum;
            m = mnew;
            #pragma unroll
            for (int ht = 0; ht < 8; ++ht) acc[ht] *= alpha;

            // ---- pack P^T to f16, build B-fragment via xor-16 exchange ----
            unsigned p0[4], p1[4];
            #pragma unroll
            for (int d = 0; d < 4; ++d) {
                union { _Float16 h2[2]; unsigned u; } t0, t1;
                t0.h2[0] = (_Float16)c0[2 * d];
                t0.h2[1] = (_Float16)c0[2 * d + 1];
                t1.h2[0] = (_Float16)c1[2 * d];
                t1.h2[1] = (_Float16)c1[2 * d + 1];
                p0[d] = t0.u;
                p1[d] = t1.u;
            }
            Frag16 bp;
            #pragma unroll
            for (int d = 0; d < 4; ++d) {
                const unsigned o0 = (unsigned)__shfl_xor((int)p0[d], 16, 32);
                const unsigned o1 = (unsigned)__shfl_xor((int)p1[d], 16, 32);
                if (!hi) { bp.u[d] = p0[d]; bp.u[4 + d] = o0; }    // s 0..15
                else     { bp.u[d] = o1;    bp.u[4 + d] = p1[d]; } // s 16..31
            }

            // ---- O^T += V^T (A, 16h x 32s) x P^T (B, 32s x 16q) ----
            #pragma unroll
            for (int ht = 0; ht < 8; ++ht) {
                Frag16 av;
                const _Float16* vp = vb + (ht * 16 + lo) * 32 + (hi ? 16 : 0);
                av.h[0] = *(const v8h*)vp;
                av.h[1] = *(const v8h*)(vp + 8);
                acc[ht] = __builtin_amdgcn_wmma_f32_16x16x32_f16(false, av.v, false, bp.v,
                                                                 (short)0, acc[ht], false, false);
            }
        }
        __syncthreads();                         // all waves done with buf i&1
    }

    // ---- epilogue: out[b][q0+lo][h] = O^T[h][q] / l ----
    const float inv = 1.0f / l;
    #pragma unroll
    for (int ht = 0; ht < 8; ++ht) {
        const v8f o = acc[ht] * inv;   // rows r -> h = ht*16 + 8*hi + r
        float* orow = out + ((size_t)(b * T_SEQ) + q0 + lo) * HD + ht * 16 + (hi ? 8 : 0);
        *(v8f*)orow = o;
    }
}

// ---------------------------------------------------------------------------
extern "C" void kernel_launch(void* const* d_in, const int* in_sizes, int n_in,
                              void* d_out, int out_size, void* d_ws, size_t ws_size,
                              hipStream_t stream)
{
    (void)in_sizes; (void)n_in; (void)out_size; (void)ws_size;

    const float* x  = (const float*)d_in[0];
    const float* Wq = (const float*)d_in[1];
    const float* Wk = (const float*)d_in[2];
    const float* Wv = (const float*)d_in[3];
    float* out = (float*)d_out;

    const size_t nTok = (size_t)B_BATCH * T_SEQ;          // 32768
    _Float16* Qh  = (_Float16*)d_ws;                      // [B][T][128]  8 MB
    _Float16* Kh  = Qh + nTok * HD;                       // [B][T][128]  8 MB
    _Float16* VTh = Kh + nTok * HD;                       // [B][128][T]  8 MB

    qkv_proj_kernel<<<(int)nTok, 128, 0, stream>>>(x, Wq, Wk, Wv, Qh, Kh, VTh);

    const int blocks = B_BATCH * (T_SEQ / 128);           // 256 blocks x 8 waves
    flash_attn_kernel<<<blocks, 256, 0, stream>>>(Qh, Kh, VTh, out);
}
